// CausalSelfAttention_11836929867801
// MI455X (gfx1250) — compile-verified, matrix operand order checked by reading, not run on hardware
//
#include <hip/hip_runtime.h>
#include <stdint.h>

// ---- problem constants (match reference) ----
#define B_    2
#define S_    2048
#define E_    1024
#define H_    16
#define D_    64
#define GATE_ 32
#define MTOT  (B_ * S_)          // 4096 token rows

typedef __attribute__((ext_vector_type(16))) __bf16 v16bf;
typedef __attribute__((ext_vector_type(8)))  float  v8f;
typedef __attribute__((ext_vector_type(4)))  uint32_t u32x4;
typedef __attribute__((ext_vector_type(8)))  int      i32x8;
typedef __attribute__((ext_vector_type(4)))  int      i32x4;

union FragBF { uint32_t u[8]; v16bf v; };

__device__ __forceinline__ uint16_t f2bf(float f) {
  uint32_t u = __float_as_uint(f);
  uint32_t r = u + 0x7FFFu + ((u >> 16) & 1u);   // round-to-nearest-even
  return (uint16_t)(r >> 16);
}

// dword offset inside a 32-element (16-dword) K-chunk for fragment reg j, lane-half hh.
// ISA 7.12.2: lanes 0-15 hold K={0..7,16..23}, lanes 16-31 hold K={8..15,24..31}.
__device__ __forceinline__ int frag_off(int j, int hh) {
  return (hh << 2) + ((j < 4) ? j : j + 4);
}

// generic -> LDS byte offset (addrspace(3) pointer value is the LDS address)
__device__ __forceinline__ uint32_t lds_off_u32(void* p) {
  return (uint32_t)(uintptr_t)(__attribute__((address_space(3))) void*)p;
}

// ---- Tensor Data Mover: 2-D tile load (bf16 elements) global -> LDS ----
// tile_w elems per row, tile_h rows, row stride (elems). CDNA5 ISA.
// 6-arg builtin form (clang-23 / therock-10.0): (g0 x4, g1 x8, g2 x4, g3 x4, x8, cpol)
__device__ __forceinline__ void tdm_load_2d(uint32_t lds_addr, const void* gptr,
                                            uint32_t tile_w, uint32_t tile_h,
                                            uint32_t stride_elems) {
  const uint64_t ga = (uint64_t)(uintptr_t)gptr;
  const uint32_t td0 = stride_elems;      // tensor_dim0 (row length for OOB; tiles in-bounds)
  const uint32_t td1 = 1u << 20;          // tensor_dim1: generous, never OOB here
  u32x4 g0;
  g0[0] = 1u;                                           // count=1, user descriptor
  g0[1] = lds_addr;                                     // LDS byte address
  g0[2] = (uint32_t)ga;                                 // global_addr[31:0]
  g0[3] = (uint32_t)((ga >> 32) & 0x01FFFFFFu) | (2u << 30);  // addr[56:32] | type=2
  i32x8 g1;
  g1[0] = (int)(1u << 16);                              // wg_mask=0, data_size=1 (2 bytes)
  g1[1] = (int)((td0 & 0xFFFFu) << 16);                 // atomic_barrier=0 | tensor_dim0[15:0]
  g1[2] = (int)((td0 >> 16) | ((td1 & 0xFFFFu) << 16)); // tensor_dim0[31:16] | tensor_dim1[15:0]
  g1[3] = (int)((td1 >> 16) | (tile_w << 16));          // tensor_dim1[31:16] | tile_dim0
  g1[4] = (int)tile_h;                                  // tile_dim1 | tile_dim2=0
  g1[5] = (int)stride_elems;                            // tensor_dim0_stride[31:0]
  g1[6] = 0;                                            // stride[47:32] | dim1_stride lo
  g1[7] = 0;
  i32x4 z4; z4[0] = 0; z4[1] = 0; z4[2] = 0; z4[3] = 0; // groups 2/3: dims >=3 unused
  i32x8 z8;
#pragma unroll
  for (int i = 0; i < 8; ++i) z8[i] = 0;
  __builtin_amdgcn_tensor_load_to_lds(g0, g1, z4, z4, z8, 0);
}

// ---------------- conversion kernels ----------------
__global__ void conv_x_kernel(const float* __restrict__ x, uint16_t* __restrict__ xb, size_t n) {
  for (size_t i = (size_t)blockIdx.x * blockDim.x + threadIdx.x; i < n;
       i += (size_t)gridDim.x * blockDim.x)
    xb[i] = f2bf(x[i]);
}

// WT[n][k] = W[k][n], fp32 -> bf16.  W is K x N row-major.
__global__ void conv_T_kernel(const float* __restrict__ W, uint16_t* __restrict__ WT,
                              int K, int N) {
  size_t tot = (size_t)K * N;
  for (size_t i = (size_t)blockIdx.x * blockDim.x + threadIdx.x; i < tot;
       i += (size_t)gridDim.x * blockDim.x) {
    int n = (int)(i / K);
    int k = (int)(i % K);
    WT[i] = f2bf(W[(size_t)k * N + n]);
  }
}

// ------- blocked bf16 WMMA GEMM with TDM double-buffered LDS staging -------
// C(f32, MxN) = A(bf16, MxK) @ BT(bf16, NxK)^T
// Workgroup: 128 threads (4 waves) -> 64x64 C macro-tile; wave w owns rows w*16..w*16+15.
// Per 32-wide K chunk: TDM stages A-tile(64x32) and B-tile(64x32) into LDS (wave 0 issues),
// each wave does 1 A-frag load + 4 B-frag loads + 4 v_wmma_f32_16x16x32_bf16.
__global__ void gemm_bf16_kernel(const uint16_t* __restrict__ A,
                                 const uint16_t* __restrict__ BT,
                                 float* __restrict__ C,
                                 int M, int N, int K) {
  __shared__ uint16_t sA[2][64 * 32];    // 2 x 4 KiB
  __shared__ uint16_t sB[2][64 * 32];    // 2 x 4 KiB

  const int tid  = threadIdx.x;          // 0..127
  const int wv   = tid >> 5;
  const int lane = tid & 31;
  const int hh   = lane >> 4;
  const int l16  = lane & 15;
  const int n0   = blockIdx.x * 64;
  const int m0   = blockIdx.y * 64;

  const uint32_t offA[2] = { lds_off_u32(&sA[0][0]), lds_off_u32(&sA[1][0]) };
  const uint32_t offB[2] = { lds_off_u32(&sB[0][0]), lds_off_u32(&sB[1][0]) };

  v8f acc[4];
#pragma unroll
  for (int nt = 0; nt < 4; ++nt) acc[nt] = (v8f){0.f,0.f,0.f,0.f,0.f,0.f,0.f,0.f};

  if (wv == 0) {                         // uniform branch: waves 1-3 jump over (TDM ignores EXEC)
    tdm_load_2d(offA[0], A  + (size_t)m0 * K, 32, 64, (uint32_t)K);
    tdm_load_2d(offB[0], BT + (size_t)n0 * K, 32, 64, (uint32_t)K);
  }

  for (int kk = 0; kk < K; kk += 32) {
    const int cur = (kk >> 5) & 1;
    if (wv == 0) {
      if (kk + 32 < K) {
        tdm_load_2d(offA[cur ^ 1], A  + (size_t)m0 * K + kk + 32, 32, 64, (uint32_t)K);
        tdm_load_2d(offB[cur ^ 1], BT + (size_t)n0 * K + kk + 32, 32, 64, (uint32_t)K);
        __builtin_amdgcn_s_wait_tensorcnt(2);   // in-order: <=2 outstanding => current pair landed
      } else {
        __builtin_amdgcn_s_wait_tensorcnt(0);
      }
    }
    __syncthreads();                     // publish staged tiles to all waves

    FragBF a;
    const uint32_t* Ar = (const uint32_t*)(sA[cur] + (wv * 16 + l16) * 32);
#pragma unroll
    for (int j = 0; j < 8; ++j) a.u[j] = Ar[frag_off(j, hh)];

#pragma unroll
    for (int nt = 0; nt < 4; ++nt) {
      FragBF b;
      const uint32_t* Br = (const uint32_t*)(sB[cur] + (nt * 16 + l16) * 32);
#pragma unroll
      for (int j = 0; j < 8; ++j) b.u[j] = Br[frag_off(j, hh)];
      acc[nt] = __builtin_amdgcn_wmma_f32_16x16x32_bf16(false, a.v, false, b.v,
                                                        (short)0, acc[nt], false, false);
    }
    __syncthreads();                     // done reading before next TDM overwrites this buffer
  }

#pragma unroll
  for (int nt = 0; nt < 4; ++nt)
#pragma unroll
    for (int r = 0; r < 8; ++r) {
      const int cm = m0 + wv * 16 + hh * 8 + r;
      const int cn = n0 + nt * 16 + l16;
      C[(size_t)cm * N + cn] = acc[nt][r];
    }
}

// ---------------- RoPE + RMSNorm for q ----------------
// q_raw: [B*S][H*D] f32  ->  q_bf: [B][H][S][D] bf16
__global__ void rope_rms_q_kernel(const float* __restrict__ qr,
                                  const float* __restrict__ cosb,
                                  const float* __restrict__ sinb,
                                  uint16_t* __restrict__ qb) {
  const int idx = blockIdx.x;           // (b*S + s)*H + h
  const int h  = idx % H_;
  const int bs = idx / H_;
  const int s  = bs % S_;
  const int b  = bs / S_;
  const int l  = threadIdx.x;           // 0..31

  const float* qrow = qr + (size_t)bs * (H_ * D_) + h * D_;
  const float x1 = qrow[l];
  const float x2 = qrow[l + 32];
  const float c  = cosb[s * 32 + l];
  const float sn = sinb[s * 32 + l];
  const float o1 =  x1 * c + x2 * sn;
  const float o2 = -x1 * sn + x2 * c;

  float ss = o1 * o1 + o2 * o2;
#pragma unroll
  for (int d = 16; d >= 1; d >>= 1) ss += __shfl_xor(ss, d, 32);
  const float r = rsqrtf(ss * (1.0f / 64.0f) + 1.1920929e-07f);

  uint16_t* orow = qb + ((size_t)(b * H_ + h) * S_ + s) * D_;
  orow[l]      = f2bf(o1 * r);
  orow[l + 32] = f2bf(o2 * r);
}

// ---------------- k RoPE+RMSNorm, gate, v update ----------------
// k_raw/v_raw: [B*S][D] f32 -> k_bf: [B][S][D] bf16 ; vT_bf: [B][D][S] bf16
__global__ void kv_post_kernel(const float* __restrict__ kr,
                               const float* __restrict__ vr,
                               const float* __restrict__ x,
                               const float* __restrict__ ve,
                               const float* __restrict__ cosb,
                               const float* __restrict__ sinb,
                               const float* __restrict__ Wg,
                               uint16_t* __restrict__ kb,
                               uint16_t* __restrict__ vTb) {
  const int bs = blockIdx.x;
  const int s  = bs % S_;
  const int b  = bs / S_;
  const int l  = threadIdx.x;

  // ---- k: rope + rmsnorm ----
  const float* krow = kr + (size_t)bs * D_;
  const float x1 = krow[l];
  const float x2 = krow[l + 32];
  const float c  = cosb[s * 32 + l];
  const float sn = sinb[s * 32 + l];
  const float o1 =  x1 * c + x2 * sn;
  const float o2 = -x1 * sn + x2 * c;
  float ss = o1 * o1 + o2 * o2;
#pragma unroll
  for (int d = 16; d >= 1; d >>= 1) ss += __shfl_xor(ss, d, 32);
  const float r = rsqrtf(ss * (1.0f / 64.0f) + 1.1920929e-07f);
  uint16_t* korow = kb + (size_t)bs * D_;
  korow[l]      = f2bf(o1 * r);
  korow[l + 32] = f2bf(o2 * r);

  // ---- gate = 2*sigmoid(x[:GATE] @ Wg), KV==1 ----
  float g = x[(size_t)bs * E_ + l] * Wg[l];
#pragma unroll
  for (int d = 16; d >= 1; d >>= 1) g += __shfl_xor(g, d, 32);
  g = 2.0f / (1.0f + __expf(-g));

  // ---- v = v_raw + gate * ve, stored transposed [B][D][S] ----
  const float* vrow  = vr + (size_t)bs * D_;
  const float* verow = ve + (size_t)bs * D_;
  const float v1 = vrow[l]      + g * verow[l];
  const float v2 = vrow[l + 32] + g * verow[l + 32];
  vTb[((size_t)b * D_ + l)      * S_ + s] = f2bf(v1);
  vTb[((size_t)b * D_ + l + 32) * S_ + s] = f2bf(v2);
}

// ---------------- sliding-window flash attention (one wave per (b,h,16-query tile)) ----------------
__global__ void attn_kernel(const uint16_t* __restrict__ qb,   // [B][H][S][D]
                            const uint16_t* __restrict__ kb,   // [B][S][D]
                            const uint16_t* __restrict__ vT,   // [B][D][S]
                            uint16_t* __restrict__ ob,         // [B][S][E] bf16
                            const int* __restrict__ winp) {
  __shared__ uint16_t lds_p[16 * 32];      // P tile staging, 1 KB

  const int win = *winp;
  int wid = blockIdx.x;                    // ((b*H)+h)*QT + qt
  const int QT = S_ / 16;
  const int qt = wid % QT;  wid /= QT;
  const int h  = wid % H_;
  const int b  = wid / H_;

  const int lane = threadIdx.x;
  const int hh   = lane >> 4;
  const int l16  = lane & 15;
  const int s0   = qt * 16;

  // preload Q A-fragments for both 32-wide K chunks of head dim 64
  const uint32_t* Qrow =
      (const uint32_t*)(qb + ((size_t)(b * H_ + h) * S_ + s0 + l16) * D_);
  FragBF qa[2];
#pragma unroll
  for (int cth = 0; cth < 2; ++cth)
#pragma unroll
    for (int j = 0; j < 8; ++j) qa[cth].u[j] = Qrow[cth * 16 + frag_off(j, hh)];

  float m_run[8], l_run[8];
  v8f o[4];
#pragma unroll
  for (int r = 0; r < 8; ++r) { m_run[r] = -1e30f; l_run[r] = 0.f; }
#pragma unroll
  for (int ot = 0; ot < 4; ++ot) o[ot] = (v8f){0.f,0.f,0.f,0.f,0.f,0.f,0.f,0.f};

  int lo = s0 - (win - 1);
  if (lo < 0) lo = 0;
  const int kc0 = lo & ~31;

  for (int kc = kc0; kc <= s0 + 15; kc += 32) {
    // ---- scores: two 16x16 tiles over this 32-key chunk ----
    float sc0[8], sc1[8];
#pragma unroll
    for (int ntile = 0; ntile < 2; ++ntile) {
      int tok = kc + ntile * 16 + l16;
      int tokc = tok < (S_ - 1) ? tok : (S_ - 1);   // clamp addr; value masked anyway
      const uint32_t* Krow = (const uint32_t*)(kb + ((size_t)b * S_ + tokc) * D_);
      v8f sacc = {0.f,0.f,0.f,0.f,0.f,0.f,0.f,0.f};
#pragma unroll
      for (int cth = 0; cth < 2; ++cth) {
        FragBF kf;
#pragma unroll
        for (int j = 0; j < 8; ++j) kf.u[j] = Krow[cth * 16 + frag_off(j, hh)];
        sacc = __builtin_amdgcn_wmma_f32_16x16x32_bf16(false, qa[cth].v, false, kf.v,
                                                       (short)0, sacc, false, false);
      }
#pragma unroll
      for (int r = 0; r < 8; ++r) {
        const int sq = s0 + hh * 8 + r;
        const int sk = kc + ntile * 16 + l16;
        const int dist = sq - sk;
        const bool ok = (dist >= 0) && (dist < win);
        const float v = sacc[r] * 0.125f;            // D^-0.5
        if (ntile == 0) sc0[r] = ok ? v : -1e30f;
        else            sc1[r] = ok ? v : -1e30f;
      }
    }

    // ---- online softmax per row, write P(bf16) into LDS ----
#pragma unroll
    for (int r = 0; r < 8; ++r) {
      float mx = fmaxf(sc0[r], sc1[r]);
#pragma unroll
      for (int d = 8; d >= 1; d >>= 1) mx = fmaxf(mx, __shfl_xor(mx, d, 32));
      const float mn = fmaxf(m_run[r], mx);
      const float corr = __expf(m_run[r] - mn);
      const float p0 = (sc0[r] > -1e29f) ? __expf(sc0[r] - mn) : 0.f;
      const float p1 = (sc1[r] > -1e29f) ? __expf(sc1[r] - mn) : 0.f;
      float ps = p0 + p1;
#pragma unroll
      for (int d = 8; d >= 1; d >>= 1) ps += __shfl_xor(ps, d, 32);
      l_run[r] = l_run[r] * corr + ps;
      m_run[r] = mn;
#pragma unroll
      for (int ot = 0; ot < 4; ++ot) o[ot][r] *= corr;
      lds_p[(hh * 8 + r) * 32 + l16]      = f2bf(p0);
      lds_p[(hh * 8 + r) * 32 + 16 + l16] = f2bf(p1);
    }
    __syncthreads();

    // ---- P (16x32) A-fragment from LDS ----
    FragBF pa;
    const uint32_t* Prow = (const uint32_t*)(lds_p + l16 * 32);
#pragma unroll
    for (int j = 0; j < 8; ++j) pa.u[j] = Prow[frag_off(j, hh)];

    // ---- O += P @ V  (V via transposed layout: contiguous token pairs) ----
#pragma unroll
    for (int ot = 0; ot < 4; ++ot) {
      const int nd = ot * 16 + l16;
      const uint32_t* Vrow = (const uint32_t*)(vT + ((size_t)b * D_ + nd) * S_ + kc);
      FragBF vf;
#pragma unroll
      for (int j = 0; j < 8; ++j) vf.u[j] = Vrow[frag_off(j, hh)];
      o[ot] = __builtin_amdgcn_wmma_f32_16x16x32_bf16(false, pa.v, false, vf.v,
                                                      (short)0, o[ot], false, false);
    }
    __syncthreads();
  }

  // ---- epilogue: normalize, write bf16 to [B][S][H*D] ----
  float invl[8];
#pragma unroll
  for (int r = 0; r < 8; ++r) invl[r] = 1.0f / l_run[r];
#pragma unroll
  for (int ot = 0; ot < 4; ++ot)
#pragma unroll
    for (int r = 0; r < 8; ++r) {
      const int row = s0 + hh * 8 + r;
      ob[((size_t)(b * S_ + row)) * E_ + h * D_ + ot * 16 + l16] = f2bf(o[ot][r] * invl[r]);
    }
}

// ---------------- host-side orchestration ----------------
extern "C" void kernel_launch(void* const* d_in, const int* in_sizes, int n_in,
                              void* d_out, int out_size, void* d_ws, size_t ws_size,
                              hipStream_t stream) {
  (void)in_sizes; (void)n_in; (void)out_size; (void)ws_size;
  const float* x    = (const float*)d_in[0];
  const float* ve   = (const float*)d_in[1];
  const float* cosb = (const float*)d_in[2];
  const float* sinb = (const float*)d_in[3];
  const float* Wq   = (const float*)d_in[4];
  const float* Wk   = (const float*)d_in[5];
  const float* Wv   = (const float*)d_in[6];
  const float* Wo   = (const float*)d_in[7];
  const float* Wg   = (const float*)d_in[8];
  const int*   winp = (const int*)d_in[9];

  uint8_t* w = (uint8_t*)d_ws;
  // workspace layout (bytes)
  uint16_t* x_bf   = (uint16_t*)(w + 0);          // 4096*1024*2  =  8 MiB
  uint16_t* WqT    = (uint16_t*)(w + 8388608);    // 1024*1024*2  =  2 MiB
  uint16_t* WkT    = (uint16_t*)(w + 10485760);   //   64*1024*2  = 128 KiB
  uint16_t* WvT    = (uint16_t*)(w + 10616832);   //   64*1024*2  = 128 KiB
  uint16_t* WoT    = (uint16_t*)(w + 10747904);   // 1024*1024*2  =  2 MiB
  float*    q_raw  = (float*)   (w + 12845056);   // 4096*1024*4  = 16 MiB
  float*    k_raw  = (float*)   (w + 29622272);   // 4096*64*4    =  1 MiB
  float*    v_raw  = (float*)   (w + 30670848);   // 4096*64*4    =  1 MiB
  uint16_t* q_bf   = (uint16_t*)(w + 31719424);   // 4096*1024*2  =  8 MiB
  uint16_t* k_bf   = (uint16_t*)(w + 40108032);   // 4096*64*2    = 512 KiB
  uint16_t* vT_bf  = (uint16_t*)(w + 40632320);   // 4096*64*2    = 512 KiB
  uint16_t* att_bf = (uint16_t*)(w + 41156608);   // 4096*1024*2  =  8 MiB (~47.25 MiB total)

  // 1) conversions
  conv_x_kernel<<<16384, 256, 0, stream>>>(x, x_bf, (size_t)MTOT * E_);
  conv_T_kernel<<<4096, 256, 0, stream>>>(Wq, WqT, E_, H_ * D_);
  conv_T_kernel<<<256,  256, 0, stream>>>(Wk, WkT, E_, D_);
  conv_T_kernel<<<256,  256, 0, stream>>>(Wv, WvT, E_, D_);
  conv_T_kernel<<<4096, 256, 0, stream>>>(Wo, WoT, E_, E_);

  // 2) projections: blocked WMMA GEMM with TDM-staged LDS tiles
  gemm_bf16_kernel<<<dim3(16, 64), 128, 0, stream>>>(x_bf, WqT, q_raw, MTOT, H_ * D_, E_);
  gemm_bf16_kernel<<<dim3(1, 64),  128, 0, stream>>>(x_bf, WkT, k_raw, MTOT, D_, E_);
  gemm_bf16_kernel<<<dim3(1, 64),  128, 0, stream>>>(x_bf, WvT, v_raw, MTOT, D_, E_);

  // 3) fp32 postprocessing
  rope_rms_q_kernel<<<MTOT * H_, 32, 0, stream>>>(q_raw, cosb, sinb, q_bf);
  kv_post_kernel<<<MTOT, 32, 0, stream>>>(k_raw, v_raw, x, ve, cosb, sinb, Wg, k_bf, vT_bf);

  // 4) sliding-window attention (WMMA for QK^T and PV)
  attn_kernel<<<B_ * H_ * (S_ / 16), 32, 0, stream>>>(q_bf, k_bf, vT_bf, att_bf, winp);

  // 5) output projection -> fp32 d_out
  gemm_bf16_kernel<<<dim3(16, 64), 128, 0, stream>>>(att_bf, WoT, (float*)d_out, MTOT, E_, E_);
}
